// PointMemoryTransformer_28363964023114
// MI455X (gfx1250) — matrix operand order repaired and checked by reading, not compiled, not run on hardware
//
#include <hip/hip_runtime.h>
#include <hip/hip_bf16.h>
#include <math.h>
#include <stdint.h>

typedef __attribute__((ext_vector_type(16))) _Float16 v16h;
typedef __attribute__((ext_vector_type(8)))  _Float16 v8h;
typedef __attribute__((ext_vector_type(8)))  float    v8f;

#define OBS 8
#define HOR 8
#define PD 128
#define HID 256
#define NH 8
#define PROTO 32
#define BB 8
#define NN 4096
#define F56 56

__device__ __forceinline__ float gelu_exact(float x){
  return 0.5f * x * (1.0f + erff(x * 0.70710678118654752f));
}
// reduce across the 16 lanes of a half-wave (lane bits 0..3); wave32 shuffle
__device__ __forceinline__ float red16(float v){
  v += __shfl_xor(v, 1, 32);
  v += __shfl_xor(v, 2, 32);
  v += __shfl_xor(v, 4, 32);
  v += __shfl_xor(v, 8, 32);
  return v;
}

// Async global->LDS copy of one 16-byte chunk (CDNA5 async path, ASYNCcnt).
// ldsoff = LDS byte offset (low 32 bits of generic shared pointer), ga = global addr.
__device__ __forceinline__ void async_copy_b128(uint32_t ldsoff, const void* gp){
  uint64_t ga = (uint64_t)(uintptr_t)gp;
  asm volatile("global_load_async_to_lds_b128 %0, %1, off"
               :: "v"(ldsoff), "v"(ga) : "memory");
}
__device__ __forceinline__ void async_wait0(){
  asm volatile("s_wait_asynccnt 0x0" ::: "memory");
}

// ---------------------------------------------------------------------------
// K_prep: transpose + f16-convert GEMM weights into column-major "B operand"
// layout wt[col*Kd + k] so a lane fetches 16 consecutive K halves in one 32B
// load (contiguous k order this code uses for both A and B: lane n -> col n, K kb..kb+15
// for lanes 0..15 and kb+16..kb+31 for lanes 16..31).
// ---------------------------------------------------------------------------
__global__ void k_prep(const float* __restrict__ enc_w1, const float* __restrict__ enc_w2,
                       const float* __restrict__ rh_w1, const float* __restrict__ vh_w1,
                       _Float16* __restrict__ ew1t, _Float16* __restrict__ ew2t,
                       _Float16* __restrict__ rw1t, _Float16* __restrict__ vw1t){
  int idx = blockIdx.x * blockDim.x + threadIdx.x;
  if (idx < 128*64){                      // enc_w1: (56,128) -> [col][k<64] (K padded)
    int col = idx >> 6, k = idx & 63;
    ew1t[idx] = (_Float16)(k < F56 ? enc_w1[k*PD + col] : 0.f);
    return;
  }
  idx -= 128*64;
  if (idx < 128*128){                     // enc_w2: (128,128) -> [col][k]
    int col = idx >> 7, k = idx & 127;
    ew2t[idx] = (_Float16)enc_w2[k*PD + col];
    return;
  }
  idx -= 128*128;
  if (idx < 256*128){                     // rh_w1 rows 256..383 -> [col<256][k<128]
    int col = idx >> 7, k = idx & 127;
    rw1t[idx] = (_Float16)rh_w1[(256+k)*HID + col];
    return;
  }
  idx -= 256*128;
  if (idx < 128*128){                     // vh_w1 rows 256..383 -> [col<128][k<128]
    int col = idx >> 7, k = idx & 127;
    vw1t[idx] = (_Float16)vh_w1[(256+k)*(HID/2) + col];
    return;
  }
}

// ---------------------------------------------------------------------------
// K_encoder: feature build + LayerNorm + 2 WMMA GEMMs -> point_tokens (f16)
// 8 waves / block, 16 tokens per wave, 128 tokens per block, 256 blocks.
// ---------------------------------------------------------------------------
__global__ __launch_bounds__(256) void k_encoder(
    const float* __restrict__ obs_points, const int* __restrict__ obs_vis,
    const float* __restrict__ rel_xy,
    const float* __restrict__ ln_w, const float* __restrict__ ln_b,
    const _Float16* __restrict__ ew1t, const float* __restrict__ enc_b1,
    const _Float16* __restrict__ ew2t, const float* __restrict__ enc_b2,
    _Float16* __restrict__ pt_h)
{
  __shared__ _Float16 xh[8][16*64];     // post-LN features, K padded to 64
  __shared__ _Float16 xh2[8][16*128];   // hidden activations
  int w = threadIdx.x >> 5, lane = threadIdx.x & 31;
  int b = blockIdx.x >> 5;
  int tokBase = (blockIdx.x & 31)*128 + w*16;

  if (lane < 16){
    int tok = tokBase + lane;
    size_t base = ((size_t)(b*NN + tok))*OBS*2;
    float f[F56];
    float rx = rel_xy[(size_t)(b*NN+tok)*2 + 0];
    float ry = rel_xy[(size_t)(b*NN+tok)*2 + 1];
    float ppx = 0.f, ppy = 0.f;
    #pragma unroll
    for (int t = 0; t < OBS; t++){
      float px = obs_points[base + t*2 + 0];
      float py = obs_points[base + t*2 + 1];
      float vx = (t == 0) ? 0.f : (px - ppx);
      float vy = (t == 0) ? 0.f : (py - ppy);
      ppx = px; ppy = py;
      float vis = (float)obs_vis[(size_t)(b*NN+tok)*OBS + t];
      f[t*7+0]=px; f[t*7+1]=py; f[t*7+2]=vx; f[t*7+3]=vy; f[t*7+4]=vis; f[t*7+5]=rx; f[t*7+6]=ry;
    }
    float mu = 0.f;
    #pragma unroll
    for (int i = 0; i < F56; i++) mu += f[i];
    mu *= (1.f/F56);
    float var = 0.f;
    #pragma unroll
    for (int i = 0; i < F56; i++){ float d = f[i]-mu; var += d*d; }
    var *= (1.f/F56);
    float inv = rsqrtf(var + 1e-5f);
    #pragma unroll
    for (int i = 0; i < F56; i++)
      xh[w][lane*64 + i] = (_Float16)((f[i]-mu)*inv*ln_w[i] + ln_b[i]);
    #pragma unroll
    for (int i = F56; i < 64; i++) xh[w][lane*64 + i] = (_Float16)0.f;
  }
  __syncthreads();

  int row = lane & 15, grp = lane >> 4;
  // GEMM1: (16 x 64) @ (64 x 128)
  v8f acc1[8] = {};
  #pragma unroll
  for (int kb = 0; kb < 64; kb += 32){
    const _Float16* ar = &xh[w][row*64];
    v8h lo = *(const v8h*)(ar + kb + grp*16);
    v8h hi = *(const v8h*)(ar + kb + grp*16 + 8);
    v16h A;
    #pragma unroll
    for (int i = 0; i < 8; i++){ A[i] = lo[i]; A[8+i] = hi[i]; }
    #pragma unroll
    for (int tile = 0; tile < 8; tile++){
      int gcol = tile*16 + row;
      v16h Bv = *(const v16h*)(ew1t + (size_t)gcol*64 + kb + grp*16);
      acc1[tile] = __builtin_amdgcn_wmma_f32_16x16x32_f16(false, A, false, Bv,
                     (short)0, acc1[tile], false, false);
    }
  }
  #pragma unroll
  for (int tile = 0; tile < 8; tile++){
    int col = tile*16 + row;
    float b1 = enc_b1[col];
    #pragma unroll
    for (int r = 0; r < 8; r++){
      int m = r + grp*8;
      xh2[w][m*128 + col] = (_Float16)gelu_exact(acc1[tile][r] + b1);
    }
  }
  __syncthreads();

  // GEMM2: (16 x 128) @ (128 x 128)
  v8f acc2[8] = {};
  #pragma unroll
  for (int kb = 0; kb < 128; kb += 32){
    const _Float16* ar = &xh2[w][row*128];
    v8h lo = *(const v8h*)(ar + kb + grp*16);
    v8h hi = *(const v8h*)(ar + kb + grp*16 + 8);
    v16h A;
    #pragma unroll
    for (int i = 0; i < 8; i++){ A[i] = lo[i]; A[8+i] = hi[i]; }
    #pragma unroll
    for (int tile = 0; tile < 8; tile++){
      int gcol = tile*16 + row;
      v16h Bv = *(const v16h*)(ew2t + (size_t)gcol*128 + kb + grp*16);
      acc2[tile] = __builtin_amdgcn_wmma_f32_16x16x32_f16(false, A, false, Bv,
                     (short)0, acc2[tile], false, false);
    }
  }
  #pragma unroll
  for (int tile = 0; tile < 8; tile++){
    int col = tile*16 + row;
    float b2 = enc_b2[col];
    #pragma unroll
    for (int r = 0; r < 8; r++){
      int m = r + grp*8;
      pt_h[((size_t)(b*NN + tokBase + m))*PD + col] = (_Float16)gelu_exact(acc2[tile][r] + b2);
    }
  }
}

// ---------------------------------------------------------------------------
// K_small: per (b,t) block -> tq, q, and folded query vector
// qk3[b,h,t,p] = (1/sqrt(32)) * sum_c (sum_d q[h*32+d]*Wk[c,h*32+d]) * query_w[p,c]
// (softmax-shift makes the bias-offset term irrelevant, so it is dropped)
// ---------------------------------------------------------------------------
__global__ __launch_bounds__(256) void k_small(
    const float* __restrict__ anchor_obs, const float* __restrict__ anchor_obs_vel,
    const float* __restrict__ time_embed, const float* __restrict__ anchor_w,
    const float* __restrict__ anchor_b,
    const float* __restrict__ in_proj_w, const float* __restrict__ in_proj_b,
    const float* __restrict__ query_w, float* __restrict__ qk3)
{
  int b = blockIdx.x >> 3, t = blockIdx.x & 7;
  __shared__ float tq[HID];
  __shared__ float q[HID];
  __shared__ float qk2[NH*HID];
  int tid = threadIdx.x;
  {
    int j = tid;
    float s = time_embed[t*HID + j] + anchor_b[j];
    #pragma unroll
    for (int c = 0; c < 32; c++){
      int tt = c >> 2, i = c & 3;
      float a = (i < 2) ? anchor_obs[(b*OBS + tt)*2 + i]
                        : anchor_obs_vel[(b*OBS + tt)*2 + (i-2)];
      s += a * anchor_w[c*HID + j];
    }
    tq[j] = s;
  }
  __syncthreads();
  {
    int j = tid;
    float s = in_proj_b[j];
    for (int c = 0; c < HID; c++) s += tq[c]*in_proj_w[c*768 + j];
    q[j] = s;
  }
  __syncthreads();
  for (int idx = tid; idx < NH*HID; idx += 256){
    int h = idx >> 8, c = idx & 255;
    float s = 0.f;
    #pragma unroll
    for (int d = 0; d < 32; d++) s += q[h*32+d]*in_proj_w[c*768 + 256 + h*32 + d];
    qk2[idx] = s;
  }
  __syncthreads();
  for (int idx = tid; idx < NH*PD; idx += 256){
    int h = idx >> 7, p = idx & 127;
    float s = 0.f;
    for (int c = 0; c < HID; c++) s += qk2[h*HID + c]*query_w[p*HID + c];
    qk3[((size_t)((b*NH + h)*HOR + t))*PD + p] = s * 0.17677669529663687f; // 1/sqrt(32)
  }
}

// ---------------------------------------------------------------------------
// K_attn: per (b,h,t) block: scores over 4096 tokens, softmax,
// ctx128 = attn @ point_tokens  (everything else folded to the query side)
// ---------------------------------------------------------------------------
__global__ __launch_bounds__(256) void k_attn(
    const _Float16* __restrict__ pt_h, const float* __restrict__ qk3,
    float* __restrict__ ctx128)
{
  int bht = blockIdx.x;                  // ((b*NH+h)*HOR + t)
  int b = bht / (NH*HOR);
  __shared__ float qs[PD];
  __shared__ float sc[NN];
  __shared__ float red[256];
  __shared__ float part[2][PD];
  int tid = threadIdx.x;
  if (tid < PD) qs[tid] = qk3[(size_t)bht*PD + tid];
  __syncthreads();
  const _Float16* ptb = pt_h + (size_t)b*NN*PD;
  float mx = -1e30f;
  for (int n = tid; n < NN; n += 256){
    if (n + 256 < NN) __builtin_prefetch(ptb + (size_t)(n+256)*PD, 0, 0);
    const _Float16* rowp = ptb + (size_t)n*PD;
    float s = 0.f;
    #pragma unroll 4
    for (int p = 0; p < PD; p++) s += qs[p]*(float)rowp[p];
    sc[n] = s;
    mx = fmaxf(mx, s);
  }
  red[tid] = mx; __syncthreads();
  for (int off = 128; off > 0; off >>= 1){
    if (tid < off) red[tid] = fmaxf(red[tid], red[tid+off]);
    __syncthreads();
  }
  mx = red[0]; __syncthreads();
  float lsum = 0.f;
  for (int n = tid; n < NN; n += 256){
    float e = __expf(sc[n] - mx);
    sc[n] = e;
    lsum += e;
  }
  red[tid] = lsum; __syncthreads();
  for (int off = 128; off > 0; off >>= 1){
    if (tid < off) red[tid] += red[tid+off];
    __syncthreads();
  }
  float inv = 1.f / red[0];
  int p = tid & 127, hf = tid >> 7;
  float acc = 0.f;
  for (int n = hf*(NN/2); n < (hf+1)*(NN/2); n++)
    acc += sc[n]*(float)ptb[(size_t)n*PD + p];
  part[hf][p] = acc;
  __syncthreads();
  if (tid < PD) ctx128[(size_t)bht*PD + tid] = (part[0][tid] + part[1][tid])*inv;
}

// ---------------------------------------------------------------------------
// K_future: per (b,t): unfold ctx -> o -> future; then the small per-(b,t)
// head precomputations A1 = future@rh_w1[:256]+rh_b1, A2 = future@vh_w1[:256]+vh_b1,
// sem_logits, anchor deltas.
// ---------------------------------------------------------------------------
__global__ __launch_bounds__(256) void k_future(
    const float* __restrict__ ctx128, const float* __restrict__ query_w,
    const float* __restrict__ query_b,
    const float* __restrict__ in_proj_w, const float* __restrict__ in_proj_b,
    const float* __restrict__ out_proj_w, const float* __restrict__ out_proj_b,
    const float* __restrict__ rh_w1, const float* __restrict__ rh_b1,
    const float* __restrict__ vh_w1, const float* __restrict__ vh_b1,
    const float* __restrict__ ah_w, const float* __restrict__ ah_b,
    const float* __restrict__ sh_w, const float* __restrict__ sh_b,
    float* __restrict__ A1, float* __restrict__ A2,
    float* __restrict__ delta, float* __restrict__ sem_out)
{
  int b = blockIdx.x >> 3, t = blockIdx.x & 7;
  __shared__ float c128[PD];
  __shared__ float ctxm[HID];
  __shared__ float o[HID];
  __shared__ float fut[HID];
  int tid = threadIdx.x;
  for (int h = 0; h < NH; h++){
    if (tid < PD) c128[tid] = ctx128[((size_t)((b*NH + h)*HOR + t))*PD + tid];
    __syncthreads();
    {
      int j = tid;
      float s = query_b[j];                       // sum(attn)==1 folds query_b in
      for (int p = 0; p < PD; p++) s += c128[p]*query_w[p*HID + j];
      ctxm[j] = s;
    }
    __syncthreads();
    if (tid < 32){
      int d = tid;
      float s = in_proj_b[512 + h*32 + d];
      for (int c = 0; c < HID; c++) s += ctxm[c]*in_proj_w[c*768 + 512 + h*32 + d];
      o[h*32 + d] = s;
    }
    __syncthreads();
  }
  {
    int j = tid;
    float s = out_proj_b[j];
    for (int c = 0; c < HID; c++) s += o[c]*out_proj_w[c*HID + j];
    fut[j] = s;
  }
  __syncthreads();
  {
    int j = tid;
    float s = rh_b1[j];
    for (int c = 0; c < HID; c++) s += fut[c]*rh_w1[c*HID + j];
    A1[(size_t)(b*HOR + t)*HID + j] = s;
  }
  if (tid < PD){
    int j = tid;
    float s = vh_b1[j];
    for (int c = 0; c < HID; c++) s += fut[c]*vh_w1[c*PD + j];
    A2[(size_t)(b*HOR + t)*PD + j] = s;
  }
  if (tid < PROTO){
    int j = tid;
    float s = sh_b[j];
    for (int c = 0; c < HID; c++) s += fut[c]*sh_w[c*PROTO + j];
    sem_out[(size_t)(b*HOR + t)*PROTO + j] = s;
  }
  if (tid < 2){
    float s = ah_b[tid];
    for (int c = 0; c < HID; c++) s += fut[c]*ah_w[c*2 + tid];
    delta[(b*HOR + t)*2 + tid] = 0.1f * tanhf(s);
  }
}

__global__ void k_cumsum(const float* __restrict__ anchor_obs,
                         const float* __restrict__ delta,
                         float* __restrict__ anchor_pred){
  int tid = threadIdx.x;
  if (tid < 16){
    int b = tid >> 1, i = tid & 1;
    float acc = anchor_obs[(b*OBS + 7)*2 + i];
    for (int t = 0; t < HOR; t++){
      acc += delta[(b*HOR + t)*2 + i];
      anchor_pred[(b*HOR + t)*2 + i] = acc;
    }
  }
}

// ---------------------------------------------------------------------------
// K_heads: dominant kernel. Per wave: 16 tokens. WMMA GEMMs
//   Bmat1 = pt @ rh_w1[256:384]  (16x256, K=128)  + rel term
//   Bmat2 = pt @ vh_w1[256:384]  (16x128, K=128)
// then for each of 8 horizons: gelu(A + Bmat) dotted with the tiny second-layer
// weights, reduced across the 16-lane half-wave with shfl_xor.
// The per-(b) A1/A2 tables are staged global->LDS with the CDNA5 async path
// (GLOBAL_LOAD_ASYNC_TO_LDS_B128, ASYNCcnt).
// ---------------------------------------------------------------------------
__global__ __launch_bounds__(256) void k_heads(
    const _Float16* __restrict__ pt_h,
    const _Float16* __restrict__ rw1t, const _Float16* __restrict__ vw1t,
    const float* __restrict__ rel_xy, const float* __restrict__ rh_w1,
    const float* __restrict__ rh_w2, const float* __restrict__ rh_b2,
    const float* __restrict__ vh_w2, const float* __restrict__ vh_b2,
    const float* __restrict__ A1g, const float* __restrict__ A2g,
    const float* __restrict__ anchor_pred,
    float* __restrict__ pred_out, float* __restrict__ vis_out)
{
  __shared__ __align__(16) float A1s[HOR*HID];
  __shared__ __align__(16) float A2s[HOR*PD];
  __shared__ float aps[HOR*2];
  __shared__ float relS[8][16][2];
  int b = blockIdx.x >> 5;
  int blkTok = (blockIdx.x & 31)*128;
  int tid = threadIdx.x;

  // ---- async global->LDS staging of A1 (8KB) and A2 (4KB) ----
  {
    const float* s1 = A1g + (size_t)b*HOR*HID;
    const float* s2 = A2g + (size_t)b*HOR*PD;
    uint32_t d1 = (uint32_t)(uintptr_t)(void*)A1s;   // LDS byte offset (addr[31:0])
    uint32_t d2 = (uint32_t)(uintptr_t)(void*)A2s;
    // A1: 512 x 16B chunks, 2 per thread
    async_copy_b128(d1 + tid*16,        s1 + tid*4);
    async_copy_b128(d1 + (tid+256)*16,  s1 + (tid+256)*4);
    // A2: 256 x 16B chunks, 1 per thread
    async_copy_b128(d2 + tid*16,        s2 + tid*4);
    async_wait0();
  }
  if (tid < HOR*2) aps[tid] = anchor_pred[b*HOR*2 + tid];
  int w = tid >> 5, lane = tid & 31;
  int waveTok = blkTok + w*16;
  if (lane < 16){
    relS[w][lane][0] = rel_xy[((size_t)(b*NN + waveTok + lane))*2 + 0];
    relS[w][lane][1] = rel_xy[((size_t)(b*NN + waveTok + lane))*2 + 1];
  }
  __syncthreads();

  int row = lane & 15, grp = lane >> 4;
  const _Float16* aBase = pt_h + ((size_t)(b*NN + waveTok))*PD;

  // -------- Phase A: residual head (pred_points) --------
  {
    v8f acc[16] = {};
    #pragma unroll
    for (int kb = 0; kb < PD; kb += 32){
      const _Float16* ar = aBase + (size_t)row*PD;
      v8h lo = *(const v8h*)(ar + kb + grp*16);
      v8h hi = *(const v8h*)(ar + kb + grp*16 + 8);
      v16h A;
      #pragma unroll
      for (int i = 0; i < 8; i++){ A[i] = lo[i]; A[8+i] = hi[i]; }
      #pragma unroll
      for (int tile = 0; tile < 16; tile++){
        int gcol = tile*16 + row;
        v16h Bv = *(const v16h*)(rw1t + (size_t)gcol*PD + kb + grp*16);
        acc[tile] = __builtin_amdgcn_wmma_f32_16x16x32_f16(false, A, false, Bv,
                      (short)0, acc[tile], false, false);
      }
    }
    float w20[16], w21[16];
    #pragma unroll
    for (int tile = 0; tile < 16; tile++){
      int col = tile*16 + row;
      float rw0 = rh_w1[384*HID + col];
      float rw1v = rh_w1[385*HID + col];
      w20[tile] = rh_w2[col*2 + 0];
      w21[tile] = rh_w2[col*2 + 1];
      #pragma unroll
      for (int r = 0; r < 8; r++){
        int m = r + grp*8;
        acc[tile][r] += relS[w][m][0]*rw0 + relS[w][m][1]*rw1v;
      }
    }
    float rb0 = rh_b2[0], rb1 = rh_b2[1];
    for (int t = 0; t < HOR; t++){
      float pr0[8] = {}, pr1[8] = {};
      #pragma unroll
      for (int tile = 0; tile < 16; tile++){
        float a1 = A1s[t*HID + tile*16 + row];
        #pragma unroll
        for (int r = 0; r < 8; r++){
          float h = gelu_exact(a1 + acc[tile][r]);
          pr0[r] += h*w20[tile];
          pr1[r] += h*w21[tile];
        }
      }
      #pragma unroll
      for (int r = 0; r < 8; r++){
        float s0 = red16(pr0[r]);
        float s1 = red16(pr1[r]);
        if (row == 0){
          int m = r + grp*8;
          int tok = waveTok + m;
          size_t oidx = (((size_t)b*NN + tok)*HOR + t)*2;
          pred_out[oidx + 0] = aps[t*2 + 0] + 0.1f*tanhf(s0 + rb0);
          pred_out[oidx + 1] = aps[t*2 + 1] + 0.1f*tanhf(s1 + rb1);
        }
      }
    }
  }

  // -------- Phase B: visibility head --------
  {
    v8f acc[8] = {};
    #pragma unroll
    for (int kb = 0; kb < PD; kb += 32){
      const _Float16* ar = aBase + (size_t)row*PD;
      v8h lo = *(const v8h*)(ar + kb + grp*16);
      v8h hi = *(const v8h*)(ar + kb + grp*16 + 8);
      v16h A;
      #pragma unroll
      for (int i = 0; i < 8; i++){ A[i] = lo[i]; A[8+i] = hi[i]; }
      #pragma unroll
      for (int tile = 0; tile < 8; tile++){
        int gcol = tile*16 + row;
        v16h Bv = *(const v16h*)(vw1t + (size_t)gcol*PD + kb + grp*16);
        acc[tile] = __builtin_amdgcn_wmma_f32_16x16x32_f16(false, A, false, Bv,
                      (short)0, acc[tile], false, false);
      }
    }
    float wv[8];
    #pragma unroll
    for (int tile = 0; tile < 8; tile++) wv[tile] = vh_w2[tile*16 + row];
    float vb = vh_b2[0];
    for (int t = 0; t < HOR; t++){
      float pv[8] = {};
      #pragma unroll
      for (int tile = 0; tile < 8; tile++){
        float a2 = A2s[t*PD + tile*16 + row];
        #pragma unroll
        for (int r = 0; r < 8; r++)
          pv[r] += gelu_exact(a2 + acc[tile][r]) * wv[tile];
      }
      #pragma unroll
      for (int r = 0; r < 8; r++){
        float s = red16(pv[r]);
        if (row == 0){
          int m = r + grp*8;
          int tok = waveTok + m;
          vis_out[((size_t)b*NN + tok)*HOR + t] = s + vb;
        }
      }
    }
  }
}

// ---------------------------------------------------------------------------
extern "C" void kernel_launch(void* const* d_in, const int* in_sizes, int n_in,
                              void* d_out, int out_size, void* d_ws, size_t ws_size,
                              hipStream_t stream){
  (void)in_sizes; (void)n_in; (void)out_size; (void)ws_size;
  const float* obs_points     = (const float*)d_in[0];
  const int*   obs_vis        = (const int*)d_in[1];
  const float* rel_xy         = (const float*)d_in[2];
  const float* anchor_obs     = (const float*)d_in[3];
  const float* anchor_obs_vel = (const float*)d_in[4];
  const float* ln_w   = (const float*)d_in[6];
  const float* ln_b   = (const float*)d_in[7];
  const float* enc_w1 = (const float*)d_in[8];
  const float* enc_b1 = (const float*)d_in[9];
  const float* enc_w2 = (const float*)d_in[10];
  const float* enc_b2 = (const float*)d_in[11];
  const float* time_embed = (const float*)d_in[12];
  const float* anchor_w   = (const float*)d_in[13];
  const float* anchor_b   = (const float*)d_in[14];
  const float* query_w    = (const float*)d_in[15];
  const float* query_b    = (const float*)d_in[16];
  const float* in_proj_w  = (const float*)d_in[17];
  const float* in_proj_b  = (const float*)d_in[18];
  const float* out_proj_w = (const float*)d_in[19];
  const float* out_proj_b = (const float*)d_in[20];
  const float* ah_w  = (const float*)d_in[21];
  const float* ah_b  = (const float*)d_in[22];
  const float* rh_w1 = (const float*)d_in[23];
  const float* rh_b1 = (const float*)d_in[24];
  const float* rh_w2 = (const float*)d_in[25];
  const float* rh_b2 = (const float*)d_in[26];
  const float* vh_w1 = (const float*)d_in[27];
  const float* vh_b1 = (const float*)d_in[28];
  const float* vh_w2 = (const float*)d_in[29];
  const float* vh_b2 = (const float*)d_in[30];
  const float* sh_w  = (const float*)d_in[31];
  const float* sh_b  = (const float*)d_in[32];

  char* ws = (char*)d_ws;
  _Float16* pt_h = (_Float16*)(ws + 0);              // 8 MB
  _Float16* ew1t = (_Float16*)(ws + 8388608);
  _Float16* ew2t = (_Float16*)(ws + 8404992);
  _Float16* rw1t = (_Float16*)(ws + 8437760);
  _Float16* vw1t = (_Float16*)(ws + 8503296);
  float* qk3   = (float*)(ws + 8536064);
  float* ctx   = (float*)(ws + 8798208);
  float* A1    = (float*)(ws + 9060352);
  float* A2    = (float*)(ws + 9125888);
  float* delta = (float*)(ws + 9158656);

  float* out = (float*)d_out;
  float* anchor_pred_out = out + 0;        // (8,8,2)
  float* pred_out        = out + 128;      // (8,4096,8,2)
  float* vis_out         = out + 524416;   // (8,4096,8)
  float* sem_out         = out + 786560;   // (8,8,32)

  k_prep<<<288, 256, 0, stream>>>(enc_w1, enc_w2, rh_w1, vh_w1, ew1t, ew2t, rw1t, vw1t);
  k_encoder<<<256, 256, 0, stream>>>(obs_points, obs_vis, rel_xy, ln_w, ln_b,
                                     ew1t, enc_b1, ew2t, enc_b2, pt_h);
  k_small<<<64, 256, 0, stream>>>(anchor_obs, anchor_obs_vel, time_embed, anchor_w,
                                  anchor_b, in_proj_w, in_proj_b, query_w, qk3);
  k_attn<<<512, 256, 0, stream>>>(pt_h, qk3, ctx);
  k_future<<<64, 256, 0, stream>>>(ctx, query_w, query_b, in_proj_w, in_proj_b,
                                   out_proj_w, out_proj_b, rh_w1, rh_b1,
                                   vh_w1, vh_b1, ah_w, ah_b, sh_w, sh_b,
                                   A1, A2, delta, sem_out);
  k_cumsum<<<1, 32, 0, stream>>>(anchor_obs, delta, anchor_pred_out);
  k_heads<<<256, 256, 0, stream>>>(pt_h, rw1t, vw1t, rel_xy, rh_w1, rh_w2, rh_b2,
                                   vh_w2, vh_b2, A1, A2, anchor_pred_out,
                                   pred_out, vis_out);
}
